// MHA_42399917146755
// MI455X (gfx1250) — matrix operand order repaired and checked by reading, not compiled, not run on hardware
//
#include <hip/hip_runtime.h>
#include <hip/hip_bf16.h>
#include <stdint.h>

#ifndef __has_builtin
#define __has_builtin(x) 0
#endif

// ---------------------------------------------------------------------------
// CDNA5 (gfx1250) wave32 WMMA MHA pipeline:
//   gemm_k<0> : x(f32) @ Wqkv -> Q(scaled)/K/V bf16, [b,h,s,hd]   (pipelined)
//   attn_k    : flash attention (segment + causal), async-LDS K staging
//   gemm_k<1> : O(bf16) @ Wo + bo -> f32 output                    (pipelined)
// ---------------------------------------------------------------------------

typedef __attribute__((ext_vector_type(16))) __bf16 v16bf;
typedef __attribute__((ext_vector_type(8)))  float  v8f;

union FragA {                // one 16x32 (or 32x16) bf16 operand: 8 VGPRs
  v16bf v;
  unsigned short u[16];
  uint4 q[2];
};
union FragC {                // one 16x16 f32 accumulator: 8 VGPRs
  v8f v;
  float f[8];
};

__device__ __forceinline__ unsigned short f2bf(float x) {
  union { float f; unsigned u; } a; a.f = x;
  unsigned r = a.u + 0x7FFFu + ((a.u >> 16) & 1u);   // round-to-nearest-even
  return (unsigned short)(r >> 16);
}

__device__ __forceinline__ uint4 pack8(float4 f0, float4 f1) {
  uint4 r;
  r.x = (unsigned)f2bf(f0.x) | ((unsigned)f2bf(f0.y) << 16);
  r.y = (unsigned)f2bf(f0.z) | ((unsigned)f2bf(f0.w) << 16);
  r.z = (unsigned)f2bf(f1.x) | ((unsigned)f2bf(f1.y) << 16);
  r.w = (unsigned)f2bf(f1.z) | ((unsigned)f2bf(f1.w) << 16);
  return r;
}

__device__ __forceinline__ v8f wmma_bf16(const FragA& a, const FragA& b, v8f c) {
  return __builtin_amdgcn_wmma_f32_16x16x32_bf16(false, a.v, false, b.v,
                                                 (short)0, c, false, false);
}

// Async global->LDS 16B copy (ASYNCcnt-tracked, no VGPR round trip).
// LDS offset = low 32 bits of the flat shared-aperture address (ISA 10.2).
__device__ __forceinline__ void async_ld_b128(void* lds, const void* g) {
  unsigned loff = (unsigned)(size_t)lds;
  unsigned long long ga = (unsigned long long)(size_t)g;
  asm volatile("global_load_async_to_lds_b128 %0, %1, off"
               :: "v"(loff), "v"(ga) : "memory");
}
__device__ __forceinline__ void wait_async0() {
#if __has_builtin(__builtin_amdgcn_s_wait_asynccnt)
  __builtin_amdgcn_s_wait_asynccnt(0);
#else
  asm volatile("s_wait_asynccnt 0" ::: "memory");
#endif
}

// ---------------------------------------------------------------------------
// Tiled GEMM: C[M,N] = A[M,K] * B[K,N] + bias. Block 128x128, BK=32,
// 8 waves x (64x32). Software pipelined: tile k+1 prefetched into registers
// while WMMAs consume tile k from LDS.
// MODE 0: A = f32 (convert on the fly); epilogue scatters bf16 Q/K/V.
// MODE 1: A = bf16; epilogue writes f32 + bias.
// ---------------------------------------------------------------------------
template <int MODE>
__global__ __launch_bounds__(256) void gemm_k(
    const void* __restrict__ Ag, const float* __restrict__ Bg,
    const float* __restrict__ bias, int M, int N, int K,
    float* __restrict__ outF,
    unsigned short* __restrict__ qO, unsigned short* __restrict__ kO,
    unsigned short* __restrict__ vO)
{
  constexpr int LA = 40;   // 80B row stride: 16B-aligned, 16 distinct bank slots
  constexpr int LB = 40;
  __shared__ unsigned short As[128 * LA];      // [m][k]
  __shared__ unsigned short Bs[128 * LB];      // transposed: [n][k]

  const int tid = threadIdx.x;
  const int lane = tid & 31, wv = tid >> 5;
  const int hlf = lane >> 4, l16 = lane & 15;
  const int m0 = blockIdx.y * 128, n0 = blockIdx.x * 128;
  const int wm = (wv & 1) * 64, wn = (wv >> 1) * 32;

  // staging coordinates (chunk = 8 bf16).
  // A: consecutive lanes -> consecutive cols (coalesced global, 64B LDS runs)
  const int ar0 = tid >> 2,          ac0 = (tid & 3) * 8;
  const int ar1 = (tid + 256) >> 2,  ac1 = (tid & 3) * 8;
  // B: consecutive lanes -> consecutive k rows (contiguous LDS scatter stores)
  const int bk0 = tid & 31,          bn0_ = (tid >> 5) * 8;
  const int bk1 = tid & 31,          bn1_ = ((tid + 256) >> 5) * 8;

  uint4 aReg[2];
  unsigned short bReg[16];

  auto loadStage = [&](int kb) {
    if (MODE == 0) {
      const float* s0 = (const float*)Ag + (size_t)(m0 + ar0) * K + (kb + ac0);
      const float* s1 = (const float*)Ag + (size_t)(m0 + ar1) * K + (kb + ac1);
      aReg[0] = pack8(*(const float4*)s0, *(const float4*)(s0 + 4));
      aReg[1] = pack8(*(const float4*)s1, *(const float4*)(s1 + 4));
    } else {
      const unsigned short* s0 =
          (const unsigned short*)Ag + (size_t)(m0 + ar0) * K + (kb + ac0);
      const unsigned short* s1 =
          (const unsigned short*)Ag + (size_t)(m0 + ar1) * K + (kb + ac1);
      aReg[0] = *(const uint4*)s0;
      aReg[1] = *(const uint4*)s1;
    }
    const float* b0 = Bg + (size_t)(kb + bk0) * N + (n0 + bn0_);
    const float* b1 = Bg + (size_t)(kb + bk1) * N + (n0 + bn1_);
    float4 f0 = *(const float4*)b0, f1 = *(const float4*)(b0 + 4);
    float4 f2 = *(const float4*)b1, f3 = *(const float4*)(b1 + 4);
    float fv[16] = {f0.x, f0.y, f0.z, f0.w, f1.x, f1.y, f1.z, f1.w,
                    f2.x, f2.y, f2.z, f2.w, f3.x, f3.y, f3.z, f3.w};
#pragma unroll
    for (int j = 0; j < 16; ++j) bReg[j] = f2bf(fv[j]);
  };
  auto storeStage = [&]() {
    *(uint4*)&As[ar0 * LA + ac0] = aReg[0];
    *(uint4*)&As[ar1 * LA + ac1] = aReg[1];
#pragma unroll
    for (int j = 0; j < 8; ++j) Bs[(bn0_ + j) * LB + bk0] = bReg[j];
#pragma unroll
    for (int j = 0; j < 8; ++j) Bs[(bn1_ + j) * LB + bk1] = bReg[8 + j];
  };

  FragC acc[4][2];
#pragma unroll
  for (int mi = 0; mi < 4; ++mi)
#pragma unroll
    for (int ni = 0; ni < 2; ++ni)
#pragma unroll
      for (int j = 0; j < 8; ++j) acc[mi][ni].f[j] = 0.f;

  loadStage(0);
  storeStage();
  __syncthreads();

  for (int kb = 0; kb < K; kb += 32) {
    const bool more = (kb + 32) < K;
    if (more) loadStage(kb + 32);    // global loads in flight during WMMAs

    FragA af[4], bfr[2];
#pragma unroll
    for (int mi = 0; mi < 4; ++mi) {
      const unsigned short* p = &As[(wm + mi * 16 + l16) * LA + hlf * 8];
      af[mi].q[0] = *(const uint4*)p;
      af[mi].q[1] = *(const uint4*)(p + 16);
    }
#pragma unroll
    for (int ni = 0; ni < 2; ++ni) {
      const unsigned short* p = &Bs[(wn + ni * 16 + l16) * LB + hlf * 8];
      bfr[ni].q[0] = *(const uint4*)p;
      bfr[ni].q[1] = *(const uint4*)(p + 16);
    }
#pragma unroll
    for (int mi = 0; mi < 4; ++mi)
#pragma unroll
      for (int ni = 0; ni < 2; ++ni)
        acc[mi][ni].v = wmma_bf16(af[mi], bfr[ni], acc[mi][ni].v);

    if (more) {
      __syncthreads();               // all waves done reading this tile
      storeStage();                  // publish next tile
      __syncthreads();
    }
  }

  // epilogue: C frag coords: row = 8*hlf + r, col = l16
#pragma unroll
  for (int mi = 0; mi < 4; ++mi) {
#pragma unroll
    for (int ni = 0; ni < 2; ++ni) {
      int col = n0 + wn + ni * 16 + l16;
      float bia = bias[col];
#pragma unroll
      for (int r = 0; r < 8; ++r) {
        int grow = m0 + wm + mi * 16 + hlf * 8 + r;
        float val = acc[mi][ni].f[r] + bia;
        if (MODE == 1) {
          outF[(size_t)grow * N + col] = val;
        } else {
          int bb = grow >> 11, s = grow & 2047;
          int which = col >> 11, d = col & 2047;
          int hh = d >> 7, dd = d & 127;
          size_t di = (((size_t)(bb * 16 + hh)) * 2048 + s) * 128 + dd;
          if (which == 0)      qO[di] = f2bf(val * 0.08838834764831845f); // 1/sqrt(128)
          else if (which == 1) kO[di] = f2bf(val);
          else                 vO[di] = f2bf(val);
        }
      }
    }
  }
}

// ---------------------------------------------------------------------------
// Flash attention: block = (b, h, 128 q-rows), 8 waves x 16 q-rows each.
// K tile staged via global_load_async_to_lds_b128; V staged with a lane-
// contiguous transpose scatter; 32-key tiles, causally pruned.
// ---------------------------------------------------------------------------
__global__ __launch_bounds__(256) void attn_k(
    const unsigned short* __restrict__ Qg, const unsigned short* __restrict__ Kg,
    const unsigned short* __restrict__ Vg, const int* __restrict__ ids,
    unsigned short* __restrict__ Og)
{
  constexpr int S = 2048, HD = 128;
  __shared__ unsigned short Ks[32 * 136];      // [key][hd]
  __shared__ unsigned short Vt[128 * 40];      // [hd][key] (transposed)
  __shared__ unsigned short Ps[8][16 * 40];    // per-wave P scratch
  __shared__ int kid[32];

  const int tid = threadIdx.x, lane = tid & 31, wv = tid >> 5;
  const int hlf = lane >> 4, l16 = lane & 15;
  const int t = blockIdx.x;
  const int qt = t & 15, hh = (t >> 4) & 15, b = t >> 8;
  const int q0 = qt * 128;
  const int qb = q0 + wv * 16;

  const size_t headOff = ((size_t)(b * 16 + hh)) * S * HD;

  FragA qf[4];
  {
    const unsigned short* qrow = Qg + headOff + (size_t)(qb + l16) * HD;
#pragma unroll
    for (int ki = 0; ki < 4; ++ki) {
      qf[ki].q[0] = *(const uint4*)(qrow + ki * 32 + hlf * 8);
      qf[ki].q[1] = *(const uint4*)(qrow + ki * 32 + 16 + hlf * 8);
    }
  }
  int qid[8];
#pragma unroll
  for (int r = 0; r < 8; ++r) qid[r] = ids[b * S + qb + hlf * 8 + r];

  float mrow[8], lrow[8];
  FragC o[8];
#pragma unroll
  for (int r = 0; r < 8; ++r) { mrow[r] = -1e30f; lrow[r] = 0.f; }
#pragma unroll
  for (int nb = 0; nb < 8; ++nb)
#pragma unroll
    for (int j = 0; j < 8; ++j) o[nb].f[j] = 0.f;

  const int kmax = q0 + 127;
  for (int kb0 = 0; kb0 <= kmax; kb0 += 32) {
    // K tile: async global->LDS (row-major, pure byte copy)
#pragma unroll
    for (int i = 0; i < 2; ++i) {
      int c = tid + i * 256;
      int key = c >> 4, d8 = (c & 15) * 8;
      async_ld_b128(&Ks[key * 136 + d8],
                    Kg + headOff + (size_t)(kb0 + key) * HD + d8);
    }
    // V tile: transpose scatter; consecutive lanes -> consecutive keys (2B runs)
#pragma unroll
    for (int i = 0; i < 2; ++i) {
      int c = tid + i * 256;
      int key = c & 31, d8 = (c >> 5) * 8;
      uint4 vvx = *(const uint4*)(Vg + headOff + (size_t)(kb0 + key) * HD + d8);
      const unsigned short* pv = (const unsigned short*)&vvx;
#pragma unroll
      for (int j = 0; j < 8; ++j) Vt[(d8 + j) * 40 + key] = pv[j];
    }
    if (tid < 32) kid[tid] = ids[b * S + kb0 + tid];
    wait_async0();
    __syncthreads();

    // scores: S(16x32) = Q(16x128) @ K^T, 8 WMMAs
    FragC s0, s1;
#pragma unroll
    for (int j = 0; j < 8; ++j) { s0.f[j] = 0.f; s1.f[j] = 0.f; }
#pragma unroll
    for (int ki = 0; ki < 4; ++ki) {
      FragA k0, k1;
      const unsigned short* p0 = &Ks[l16 * 136 + ki * 32 + hlf * 8];
      k0.q[0] = *(const uint4*)p0;  k0.q[1] = *(const uint4*)(p0 + 16);
      const unsigned short* p1 = &Ks[(16 + l16) * 136 + ki * 32 + hlf * 8];
      k1.q[0] = *(const uint4*)p1;  k1.q[1] = *(const uint4*)(p1 + 16);
      s0.v = wmma_bf16(qf[ki], k0, s0.v);
      s1.v = wmma_bf16(qf[ki], k1, s1.v);
    }

    // mask + online softmax (16-lane group reductions)
    const int kc0 = kb0 + l16, kc1 = kb0 + 16 + l16;
    const int sid0 = kid[l16], sid1 = kid[16 + l16];
#pragma unroll
    for (int r = 0; r < 8; ++r) {
      int qi = qb + hlf * 8 + r;
      bool v0 = (kc0 <= qi) && (sid0 == qid[r]);
      bool v1 = (kc1 <= qi) && (sid1 == qid[r]);
      float a0 = v0 ? s0.f[r] : -1e30f;
      float a1 = v1 ? s1.f[r] : -1e30f;
      float mx = fmaxf(a0, a1);
#pragma unroll
      for (int off = 1; off < 16; off <<= 1) mx = fmaxf(mx, __shfl_xor(mx, off, 32));
      float mnew = fmaxf(mrow[r], mx);
      float alpha = __expf(mrow[r] - mnew);        // finite sentinel -> no NaN
      float p0 = v0 ? __expf(a0 - mnew) : 0.f;
      float p1 = v1 ? __expf(a1 - mnew) : 0.f;
      float sum = p0 + p1;
#pragma unroll
      for (int off = 1; off < 16; off <<= 1) sum += __shfl_xor(sum, off, 32);
      lrow[r] = lrow[r] * alpha + sum;
      mrow[r] = mnew;
#pragma unroll
      for (int nb = 0; nb < 8; ++nb) o[nb].f[r] *= alpha;
      Ps[wv][(hlf * 8 + r) * 40 + l16]      = f2bf(p0);
      Ps[wv][(hlf * 8 + r) * 40 + 16 + l16] = f2bf(p1);
    }
    // same-wave LDS RAW (C-layout -> A-layout transpose through Ps)
    asm volatile("s_wait_dscnt 0" ::: "memory");

    // O += P(16x32) @ V(32x128), 8 WMMAs
    FragA pf;
    {
      const unsigned short* pp = &Ps[wv][l16 * 40 + hlf * 8];
      pf.q[0] = *(const uint4*)pp;
      pf.q[1] = *(const uint4*)(pp + 16);
    }
#pragma unroll
    for (int nb = 0; nb < 8; ++nb) {
      FragA vf;
      const unsigned short* vp = &Vt[(nb * 16 + l16) * 40 + hlf * 8];
      vf.q[0] = *(const uint4*)vp;
      vf.q[1] = *(const uint4*)(vp + 16);
      o[nb].v = wmma_bf16(pf, vf, o[nb].v);
    }
    __syncthreads();
  }

  // normalize + write O as bf16 in [b, s, h*hd] for the out-proj GEMM
  float inv[8];
#pragma unroll
  for (int r = 0; r < 8; ++r) inv[r] = (lrow[r] > 0.f) ? (1.f / lrow[r]) : 0.f;
#pragma unroll
  for (int nb = 0; nb < 8; ++nb)
#pragma unroll
    for (int r = 0; r < 8; ++r) {
      int qi = qb + hlf * 8 + r;
      Og[((size_t)(b * S + qi)) * 2048 + hh * 128 + nb * 16 + l16] =
          f2bf(o[nb].f[r] * inv[r]);
    }
}

// ---------------------------------------------------------------------------
extern "C" void kernel_launch(void* const* d_in, const int* in_sizes, int n_in,
                              void* d_out, int out_size, void* d_ws, size_t ws_size,
                              hipStream_t stream) {
  (void)in_sizes; (void)n_in; (void)out_size; (void)ws_size;
  const float* x    = (const float*)d_in[0];   // [4,2048,2048] f32
  const int*   ids  = (const int*)  d_in[1];   // [4,2048]
  const float* Wqkv = (const float*)d_in[2];   // [2048,6144]
  const float* bqkv = (const float*)d_in[3];   // [6144]
  const float* Wo   = (const float*)d_in[4];   // [2048,2048]
  const float* bo   = (const float*)d_in[5];   // [2048]
  float* out = (float*)d_out;                  // [4,2048,2048] f32

  // workspace: Q,K,V [b,h,s,hd] bf16 + O [b,s,dim] bf16 = 128 MiB
  const size_t HE = (size_t)4 * 16 * 2048 * 128;
  unsigned short* qbf = (unsigned short*)d_ws;
  unsigned short* kbf = qbf + HE;
  unsigned short* vbf = kbf + HE;
  unsigned short* obf = vbf + HE;

  dim3 blk(256);
  gemm_k<0><<<dim3(6144 / 128, 8192 / 128), blk, 0, stream>>>(
      x, Wqkv, bqkv, 8192, 6144, 2048, nullptr, qbf, kbf, vbf);
  attn_k<<<dim3(4 * 16 * 16), blk, 0, stream>>>(qbf, kbf, vbf, ids, obf);
  gemm_k<1><<<dim3(2048 / 128, 8192 / 128), blk, 0, stream>>>(
      obf, Wo, bo, 8192, 2048, 2048, out, nullptr, nullptr, nullptr);
}
